// Encoder_27986006901274
// MI455X (gfx1250) — matrix operand order repaired and checked by reading, not compiled, not run on hardware
//
#include <hip/hip_runtime.h>
#include <stdint.h>

// ---------------- problem constants ----------------
#define CN    20000            // nodes
#define CE    4000             // hyperedges
#define CNNZ  120000           // incidences
#define CSTOK 8                // tokens per node/edge
#define CCS   32               // claim tokens
#define CD    768              // model dim
#define CH    4                // heads
#define CB    8                // graphs
#define CDH   192              // head dim
#define CM1   (CE + CN)        // 24000 (hyperedge space incl. self nodes)
#define CTOTE (CNNZ + CN)      // 140000 incidences incl. self loops

typedef __attribute__((ext_vector_type(16))) __bf16 v16bf;
typedef __attribute__((ext_vector_type(8)))  float  v8f;
typedef __attribute__((ext_vector_type(4)))  int    v4i;

// ---- CDNA5 async global->LDS path (guarded; falls back to sync staging) ----
// Builtin signature (from compiler diagnostic): (v4i as1*, v4i as3*, Ii off, Ii cpol)
#if defined(__gfx1250__) && __has_builtin(__builtin_amdgcn_global_load_async_to_lds_b128)
#define USE_ASYNC 1
#define ASYNC_LOAD_B128(g, l)                                              \
  __builtin_amdgcn_global_load_async_to_lds_b128(                          \
      (__attribute__((address_space(1))) v4i*)(v4i*)(void*)(g),            \
      (__attribute__((address_space(3))) v4i*)(v4i*)(void*)(l), 0, 0)
#if __has_builtin(__builtin_amdgcn_s_wait_asynccnt)
#define WAIT_ASYNC(n) __builtin_amdgcn_s_wait_asynccnt(n)
#else
#define WAIT_ASYNC(n) asm volatile("s_wait_asynccnt %0" ::"i"(n))
#endif
#else
#define USE_ASYNC 0
#define WAIT_ASYNC(n)
#endif

// ---------------- small device helpers ----------------
__device__ __forceinline__ unsigned short f2bf(float f) {
  unsigned u = __float_as_uint(f);
  unsigned r = u + 0x7fffu + ((u >> 16) & 1u);   // round-to-nearest-even
  return (unsigned short)(r >> 16);
}
// monotonic float<->uint mapping for atomicMax-based segment max
__device__ __forceinline__ unsigned fkey(float f) {
  unsigned u = __float_as_uint(f);
  return (u & 0x80000000u) ? ~u : (u | 0x80000000u);
}
__device__ __forceinline__ float funkey(unsigned k) {
  unsigned u = (k & 0x80000000u) ? (k & 0x7fffffffu) : ~k;
  return __uint_as_float(u);
}
// block (256 threads) layernorm stats over 768 values (3 per thread)
__device__ __forceinline__ void ln_stats(const float x[3], float* sm,
                                         float& mean, float& rstd) {
  int tid = threadIdx.x;
  float s = x[0] + x[1] + x[2];
  sm[tid] = s; __syncthreads();
  for (int k = 128; k > 0; k >>= 1) { if (tid < k) sm[tid] += sm[tid + k]; __syncthreads(); }
  mean = sm[0] * (1.0f / (float)CD); __syncthreads();
  float d0 = x[0] - mean, d1 = x[1] - mean, d2 = x[2] - mean;
  sm[tid] = d0 * d0 + d1 * d1 + d2 * d2; __syncthreads();
  for (int k = 128; k > 0; k >>= 1) { if (tid < k) sm[tid] += sm[tid + k]; __syncthreads(); }
  rstd = rsqrtf(sm[0] * (1.0f / (float)CD) + 1e-12f);
  __syncthreads();
}

// ---------------- f32 [K][N] -> bf16 transposed [N][K] ----------------
__global__ __launch_bounds__(256)
void cvt_t_kernel(const float* __restrict__ in, unsigned short* __restrict__ out,
                  int K, int N) {
  int idx = blockIdx.x * 256 + threadIdx.x;
  if (idx >= K * N) return;
  int k = idx / N, n = idx - k * N;
  out[(size_t)n * K + k] = f2bf(in[idx]);
}

// ---------------- token pooling + LayerNorm ----------------
__global__ __launch_bounds__(256)
void pool_ln_kernel(const int* __restrict__ ids, int S,
                    const float* __restrict__ tok,
                    const float* __restrict__ g, const float* __restrict__ b,
                    float* __restrict__ outf, unsigned short* __restrict__ outbf) {
  __shared__ float sm[256];
  int row = blockIdx.x;
  const int* rid = ids + (size_t)row * S;
  float acc[3] = {0.f, 0.f, 0.f};
  int cnt = 0;
  for (int s = 0; s < S; s++) {
    int id = rid[s];
    if (id != 0) cnt++;
    const float* t = tok + (size_t)id * CD;
    acc[0] += t[threadIdx.x];
    acc[1] += t[threadIdx.x + 256];
    acc[2] += t[threadIdx.x + 512];
  }
  float inv = 1.0f / (float)cnt;   // ids are >=1 in this workload
  float x[3] = {acc[0] * inv, acc[1] * inv, acc[2] * inv};
  float mean, rstd;
  ln_stats(x, sm, mean, rstd);
  for (int j = 0; j < 3; j++) {
    int d = threadIdx.x + 256 * j;
    float y = (x[j] - mean) * rstd * g[d] + b[d];
    outf[(size_t)row * CD + d] = y;
    if (outbf) outbf[(size_t)row * CD + d] = f2bf(y);
  }
}

// ---- bf16 WMMA GEMM: C[M,768] = act(A[M,K] @ Wt[768,K]^T (+bias)(+C)) ----
// Wt is the weight matrix stored TRANSPOSED ([N][K] row-major) so both A and B
// tiles are plain row-major copies -> async global->LDS DMA works for both.
// Double-buffered LDS pipeline, 8 waves (2x4), 4x2 16x16 tiles per wave.
__global__ __launch_bounds__(256)
void gemm_bf16_kernel(const unsigned short* __restrict__ A,
                      const unsigned short* __restrict__ Wt,
                      const float* __restrict__ bias,
                      float* __restrict__ C,
                      unsigned short* __restrict__ Cbf,
                      int M, int K, int Nc, int accumulate, int relu) {
  constexpr int SA = 40;                             // padded LDS stride (elems)
  __shared__ __attribute__((aligned(16))) unsigned short lA[2][128 * SA];
  __shared__ __attribute__((aligned(16))) unsigned short lB[2][128 * SA];
  int tid = threadIdx.x, lane = tid & 31, wid = tid >> 5;
  int wm = wid >> 2, wn = wid & 3;                   // 2 (m) x 4 (n) waves
  int bm = blockIdx.y * 128, bn = blockIdx.x * 128;
  int hf = lane >> 4, l16 = lane & 15;

  v8f acc[4][2];
  for (int i = 0; i < 4; i++)
    for (int j = 0; j < 2; j++)
      for (int r = 0; r < 8; r++) acc[i][j][r] = 0.f;

  // staging geometry: 512 16B-chunks per 128x32 tile; this thread owns chunk
  // tid (row r0) and chunk tid+256 (row r0+64). A rows are CLAMPED to M-1:
  // a garbage row only feeds the (masked) C row of the same index.
  int r0  = tid >> 2;
  int c8  = (tid & 3) << 3;
  int gm0 = bm + r0;      if (gm0 > M - 1) gm0 = M - 1;
  int gm1 = bm + r0 + 64; if (gm1 > M - 1) gm1 = M - 1;
  const unsigned short* gA0 = A  + (size_t)gm0 * K + c8;
  const unsigned short* gA1 = A  + (size_t)gm1 * K + c8;
  const unsigned short* gB0 = Wt + (size_t)(bn + r0) * K + c8;
  const unsigned short* gB1 = Wt + (size_t)(bn + r0 + 64) * K + c8;

  auto stage = [&](int kk, int buf) {
#if USE_ASYNC
    ASYNC_LOAD_B128(gA0 + kk, &lA[buf][r0 * SA + c8]);
    ASYNC_LOAD_B128(gA1 + kk, &lA[buf][(r0 + 64) * SA + c8]);
    ASYNC_LOAD_B128(gB0 + kk, &lB[buf][r0 * SA + c8]);
    ASYNC_LOAD_B128(gB1 + kk, &lB[buf][(r0 + 64) * SA + c8]);
#else
    *(uint4*)&lA[buf][r0 * SA + c8]        = *(const uint4*)(gA0 + kk);
    *(uint4*)&lA[buf][(r0 + 64) * SA + c8] = *(const uint4*)(gA1 + kk);
    *(uint4*)&lB[buf][r0 * SA + c8]        = *(const uint4*)(gB0 + kk);
    *(uint4*)&lB[buf][(r0 + 64) * SA + c8] = *(const uint4*)(gB1 + kk);
#endif
  };

  auto compute = [&](int buf) {
    // B frags: lane holds column n = n0+lane%16, K = 16*(lane/16)+0..15
    v16bf bfr[2];
    for (int j = 0; j < 2; j++) {
      int n = wn * 32 + j * 16 + l16;
      union { uint4 u[2]; v16bf v; } t2;
      t2.u[0] = *(const uint4*)&lB[buf][n * SA + hf * 16];
      t2.u[1] = *(const uint4*)&lB[buf][n * SA + hf * 16 + 8];
      bfr[j] = t2.v;
    }
    // A frags: lane row m0+lane%16, K = 16*(lane/16)+0..15
    for (int i = 0; i < 4; i++) {
      int m0 = wm * 64 + i * 16;
      union { uint4 u[2]; v16bf v; } t1;
      t1.u[0] = *(const uint4*)&lA[buf][(m0 + l16) * SA + hf * 16];
      t1.u[1] = *(const uint4*)&lA[buf][(m0 + l16) * SA + hf * 16 + 8];
      for (int j = 0; j < 2; j++)
        acc[i][j] = __builtin_amdgcn_wmma_f32_16x16x32_bf16(
            false, t1.v, false, bfr[j], (short)0, acc[i][j], false, false);
    }
  };

  int niter = K >> 5;
  stage(0, 0);
  for (int it = 0; it + 1 < niter; ++it) {
    stage((it + 1) << 5, (it + 1) & 1);  // next tile in flight
    WAIT_ASYNC(4);                       // current tile landed (in-order async)
    __syncthreads();
    compute(it & 1);
    __syncthreads();                     // done reading before it+1 overwrites
  }
  WAIT_ASYNC(0);
  __syncthreads();
  compute((niter - 1) & 1);

  // epilogue: C layout vgpr r -> m = r + 8*(lane/16), n = lane%16
  for (int i = 0; i < 4; i++)
    for (int j = 0; j < 2; j++) {
      int n = bn + wn * 32 + j * 16 + l16;
      for (int r = 0; r < 8; r++) {
        int m = bm + wm * 64 + i * 16 + r + 8 * hf;
        if (m < M) {
          float v = acc[i][j][r];
          if (bias) v += bias[n];
          if (accumulate) v += C[(size_t)m * Nc + n];
          if (relu) v = v > 0.f ? v : 0.f;
          C[(size_t)m * Nc + n] = v;
          if (Cbf) Cbf[(size_t)m * Nc + n] = f2bf(v);
        }
      }
    }
}

// ---------------- per-row head scores: leakyrelu(K . att_r / sqrt(DH)) ----------------
__global__ __launch_bounds__(256)
void kr_kernel(const float* __restrict__ Km, const float* __restrict__ att,
               float* __restrict__ out, int rows) {
  int idx = blockIdx.x * 256 + threadIdx.x;
  if (idx >= rows * CH) return;
  int i = idx >> 2, h = idx & 3;
  const float* kp = Km + (size_t)i * CD + h * CDH;
  const float* ap = att + h * CDH;
  float s = 0.f;
  for (int d = 0; d < CDH; d++) s += kp[d] * ap[d];
  s *= 0.07216878364870323f;                    // 1/sqrt(192)
  out[idx] = s > 0.f ? s : 0.2f * s;            // leaky_relu(0.2)
}

// ---------------- segment max over destinations ----------------
__global__ __launch_bounds__(256)
void seg_max_kernel(const int* __restrict__ ei0, const int* __restrict__ ei1,
                    const float* __restrict__ a, unsigned* __restrict__ amax, int e2v) {
  int idx = blockIdx.x * 256 + threadIdx.x;
  if (idx >= CTOTE * CH) return;
  int e = idx >> 2, h = idx & 3;
  int a0, a1;
  if (e < CNNZ) { a0 = ei0[e]; a1 = ei1[e]; }
  else          { a0 = e - CNNZ; a1 = CE + (e - CNNZ); }
  int g = e2v ? a1 : a0;
  int s = e2v ? a0 : a1;
  atomicMax(&amax[(size_t)s * CH + h], fkey(a[(size_t)g * CH + h]));
}

// ---------------- softmax numerator/denominator scatter ----------------
__global__ __launch_bounds__(256)
void seg_sum_kernel(const int* __restrict__ ei0, const int* __restrict__ ei1,
                    const float* __restrict__ a, const unsigned* __restrict__ amax,
                    const float* __restrict__ Vm,
                    float* __restrict__ den, float* __restrict__ num, int e2v) {
  __shared__ float ex[CH];
  __shared__ int sg, ss;
  int e = blockIdx.x;
  if (threadIdx.x == 0) {
    int a0, a1;
    if (e < CNNZ) { a0 = ei0[e]; a1 = ei1[e]; }
    else          { a0 = e - CNNZ; a1 = CE + (e - CNNZ); }
    sg = e2v ? a1 : a0;
    ss = e2v ? a0 : a1;
  }
  __syncthreads();
  int g = sg, s = ss;
  if (threadIdx.x < CH) {
    float m = funkey(amax[(size_t)s * CH + threadIdx.x]);
    if (!(m > -1e37f && m < 1e37f)) m = 0.f;    // empty-segment / NaN guard
    float v = __expf(a[(size_t)g * CH + threadIdx.x] - m);
    ex[threadIdx.x] = v;
    atomicAdd(&den[(size_t)s * CH + threadIdx.x], v);
  }
  __syncthreads();
  const float* vp = Vm + (size_t)g * CD;
  float* np = num + (size_t)s * CD;
  for (int j = 0; j < 3; j++) {
    int d = threadIdx.x + 256 * j;
    atomicAdd(&np[d], ex[d / CDH] * vp[d]);
  }
}

// ---------------- out = LN0( num/(den+eps) + att_r ) ----------------
__global__ __launch_bounds__(256)
void ln0_kernel(const float* __restrict__ num, const float* __restrict__ den,
                const float* __restrict__ att,
                const float* __restrict__ g, const float* __restrict__ b,
                float* __restrict__ outf, unsigned short* __restrict__ outbf) {
  __shared__ float sm[256];
  int row = blockIdx.x;
  float x[3];
  for (int j = 0; j < 3; j++) {
    int d = threadIdx.x + 256 * j;
    int h = d / CDH;
    x[j] = num[(size_t)row * CD + d] / (den[(size_t)row * CH + h] + 1e-16f) + att[d];
  }
  float mean, rstd;
  ln_stats(x, sm, mean, rstd);
  for (int j = 0; j < 3; j++) {
    int d = threadIdx.x + 256 * j;
    float y = (x[j] - mean) * rstd * g[d] + b[d];
    outf[(size_t)row * CD + d] = y;
    outbf[(size_t)row * CD + d] = f2bf(y);
  }
}

// ---------------- out = (relu?) LN1( aout + ff ) ----------------
__global__ __launch_bounds__(256)
void ln1_kernel(const float* __restrict__ aout, const float* __restrict__ ff,
                const float* __restrict__ g, const float* __restrict__ b,
                float* __restrict__ outf, unsigned short* __restrict__ outbf, int relu) {
  __shared__ float sm[256];
  int row = blockIdx.x;
  float x[3];
  for (int j = 0; j < 3; j++) {
    int d = threadIdx.x + 256 * j;
    x[j] = aout[(size_t)row * CD + d] + ff[(size_t)row * CD + d];
  }
  float mean, rstd;
  ln_stats(x, sm, mean, rstd);
  for (int j = 0; j < 3; j++) {
    int d = threadIdx.x + 256 * j;
    float y = (x[j] - mean) * rstd * g[d] + b[d];
    if (relu) y = y > 0.f ? y : 0.f;
    if (outf)  outf[(size_t)row * CD + d] = y;
    if (outbf) outbf[(size_t)row * CD + d] = f2bf(y);
  }
}

// ---------------- E2V input: bf16( T + claim[graph(row)] ) ----------------
__global__ __launch_bounds__(256)
void addinstr_kernel(const float* __restrict__ T, const float* __restrict__ claim,
                     const int* __restrict__ e2g, const int* __restrict__ n2g,
                     unsigned short* __restrict__ outbf) {
  int idx = blockIdx.x * 256 + threadIdx.x;
  if (idx >= CM1 * CD) return;
  int r = idx / CD, d = idx - r * CD;
  int gidx = (r < CE) ? e2g[r] : n2g[r - CE];
  outbf[idx] = f2bf(T[idx] + claim[(size_t)gidx * CD + d]);
}

// ======================= host orchestration =======================
// Input flattening assumption (jax pytree: sorted dict keys, lists in order):
//   params.layers[l] at leaf base l*24:
//     +0 E2V.Wff +1 E2V.Wk +2 E2V.Wv +3 E2V.att_r +4 E2V.bff +5 E2V.bk +6 E2V.bv
//     +7 E2V.ln0_b +8 E2V.ln0_g +9 E2V.ln1_b +10 E2V.ln1_g
//     +11..21 V2E (same sub-order)  +22 fuse_W  +23 fuse_b
//   48 ln_b, 49 ln_g, 50 tok, 51 x_s, 52 x_t, 53 pos_claim,
//   54 edge_index, 55 node2graph, 56 edge2graph
extern "C" void kernel_launch(void* const* d_in, const int* in_sizes, int n_in,
                              void* d_out, int out_size, void* d_ws, size_t ws_size,
                              hipStream_t stream) {
  (void)in_sizes; (void)n_in; (void)out_size; (void)ws_size;
  auto F = [&](int i) -> const float* { return (const float*)d_in[i]; };
  auto I = [&](int i) -> const int*   { return (const int*)d_in[i]; };

  const float* ln_b = F(48);
  const float* ln_g = F(49);
  const float* tok  = F(50);
  const int* x_s = I(51);
  const int* x_t = I(52);
  const int* pos_claim = I(53);
  const int* ei0 = I(54);
  const int* ei1 = I(54) + CNNZ;
  const int* n2g = I(55);
  const int* e2g = I(56);

  // workspace arena
  char* base = (char*)d_ws;
  size_t off = 0;
  auto alloc = [&](size_t bytes) -> void* {
    void* p = base + off;
    off += (bytes + 255) & ~(size_t)255;
    return p;
  };
  float*          S_f32   = (float*)alloc((size_t)CN * CD * 4);
  unsigned short* S_bf    = (unsigned short*)alloc((size_t)CN * CD * 2);
  float*          Ta      = (float*)alloc((size_t)CM1 * CD * 4);
  unsigned short* Tabf    = (unsigned short*)alloc((size_t)CM1 * CD * 2);
  float*          Tb      = (float*)alloc((size_t)CM1 * CD * 4);
  unsigned short* Tbbf    = (unsigned short*)alloc((size_t)CM1 * CD * 2);
  float*          claim   = (float*)alloc((size_t)CB * CD * 4);
  unsigned short* ttem_bf = (unsigned short*)alloc((size_t)CM1 * CD * 2); // also E2V input
  float*          Kbuf    = (float*)alloc((size_t)CM1 * CD * 4);          // aliased as num
  float*          Vbuf    = (float*)alloc((size_t)CM1 * CD * 4);          // aliased as ff
  float*          krb     = (float*)alloc((size_t)CM1 * CH * 4);
  unsigned*       amax    = (unsigned*)alloc((size_t)CM1 * CH * 4);
  float*          den     = (float*)alloc((size_t)CM1 * CH * 4);
  float*          aout    = (float*)alloc((size_t)CM1 * CD * 4);
  unsigned short* aoutbf  = (unsigned short*)alloc((size_t)CM1 * CD * 2);
  unsigned short* Wpool   = (unsigned short*)alloc((size_t)16 * CD * CD * 2);
  float* num = Kbuf;
  float* ffb = Vbuf;

  // transpose-convert a [768][768] f32 block into bf16 [N][K]
  auto cvt_t = [&](const float* src, unsigned short* dst) {
    cvt_t_kernel<<<(CD * CD + 255) / 256, 256, 0, stream>>>(src, dst, CD, CD);
  };
  auto gemm = [&](const unsigned short* A, const unsigned short* Wt, const float* bias,
                  float* Cp, unsigned short* Cbf, int M, int Kd, int accum, int relu) {
    dim3 grid(CD / 128, (M + 127) / 128);
    gemm_bf16_kernel<<<grid, 256, 0, stream>>>(A, Wt, bias, Cp, Cbf, M, Kd, CD, accum, relu);
  };

  // weights -> bf16 transposed (every call; deterministic)
  const size_t WS = (size_t)CD * CD;
  for (int l = 0; l < 2; l++) {
    int LB = l * 24;
    unsigned short* wl = Wpool + (size_t)l * 8 * WS;
    cvt_t(F(LB + 1),       wl + 0 * WS);   // E2V.Wk^T
    cvt_t(F(LB + 2),       wl + 1 * WS);   // E2V.Wv^T
    cvt_t(F(LB + 0),       wl + 2 * WS);   // E2V.Wff^T
    cvt_t(F(LB + 12),      wl + 3 * WS);   // V2E.Wk^T
    cvt_t(F(LB + 13),      wl + 4 * WS);   // V2E.Wv^T
    cvt_t(F(LB + 11),      wl + 5 * WS);   // V2E.Wff^T
    cvt_t(F(LB + 22),      wl + 6 * WS);   // fuse_W rows   0..767 ^T
    cvt_t(F(LB + 22) + WS, wl + 7 * WS);   // fuse_W rows 768..1535 ^T
  }

  // embeddings: pool tokens + LN (f32 + bf16)
  pool_ln_kernel<<<CN, 256, 0, stream>>>(x_s, CSTOK, tok, ln_g, ln_b, S_f32, S_bf);
  pool_ln_kernel<<<CE, 256, 0, stream>>>(x_t, CSTOK, tok, ln_g, ln_b, Ta, Tabf);
  pool_ln_kernel<<<CB, 256, 0, stream>>>(pos_claim, CCS, tok, ln_g, ln_b, claim,
                                         (unsigned short*)nullptr);
  // emb_t = concat(emb_t_E, emb_s)
  (void)hipMemcpyAsync(Ta + (size_t)CE * CD, S_f32, (size_t)CN * CD * 4,
                       hipMemcpyDeviceToDevice, stream);
  (void)hipMemcpyAsync(Tabf + (size_t)CE * CD, S_bf, (size_t)CN * CD * 2,
                       hipMemcpyDeviceToDevice, stream);

  float* T = Ta;  unsigned short* Tbf = Tabf;
  float* Tn = Tb; unsigned short* Tnbf = Tbbf;

  for (int l = 0; l < 2; l++) {
    int LB = l * 24;
    unsigned short* wl = Wpool + (size_t)l * 8 * WS;

    // ---------- V2E: nodes -> hyperedge space (CM1 targets) ----------
    gemm(S_bf, wl + 3 * WS, F(LB + 16), Kbuf, nullptr, CN, CD, 0, 0);   // K = S@Wk+bk
    gemm(S_bf, wl + 4 * WS, F(LB + 17), Vbuf, nullptr, CN, CD, 0, 0);   // V = S@Wv+bv
    kr_kernel<<<(CN * CH + 255) / 256, 256, 0, stream>>>(Kbuf, F(LB + 14), krb, CN);
    (void)hipMemsetAsync(amax, 0, (size_t)CM1 * CH * 4, stream);
    (void)hipMemsetAsync(den,  0, (size_t)CM1 * CH * 4, stream);
    (void)hipMemsetAsync(num,  0, (size_t)CM1 * CD * 4, stream);
    seg_max_kernel<<<(CTOTE * CH + 255) / 256, 256, 0, stream>>>(ei0, ei1, krb, amax, 0);
    seg_sum_kernel<<<CTOTE, 256, 0, stream>>>(ei0, ei1, krb, amax, Vbuf, den, num, 0);
    ln0_kernel<<<CM1, 256, 0, stream>>>(num, den, F(LB + 14), F(LB + 19), F(LB + 18),
                                        aout, aoutbf);
    gemm(aoutbf, wl + 5 * WS, F(LB + 15), ffb, nullptr, CM1, CD, 0, 1); // relu(out@Wff+bff)
    ln1_kernel<<<CM1, 256, 0, stream>>>(aout, ffb, F(LB + 21), F(LB + 20),
                                        nullptr, ttem_bf, 1);           // t_tem (bf16)

    // ---------- fuse: T_new = [T, t_tem] @ fuse_W + fuse_b ----------
    gemm(Tbf, wl + 6 * WS, F(LB + 23), Tn, nullptr, CM1, CD, 0, 0);
    gemm(ttem_bf, wl + 7 * WS, nullptr, Tn, Tnbf, CM1, CD, 1, 0);       // accumulate
    { float* t = T; T = Tn; Tn = t; unsigned short* u = Tbf; Tbf = Tnbf; Tnbf = u; }

    // ---------- E2V: hyperedge space -> nodes, with instruct ----------
    addinstr_kernel<<<(CM1 * CD + 255) / 256, 256, 0, stream>>>(T, claim, e2g, n2g, ttem_bf);
    gemm(ttem_bf, wl + 0 * WS, F(LB + 5), Kbuf, nullptr, CM1, CD, 0, 0);
    gemm(ttem_bf, wl + 1 * WS, F(LB + 6), Vbuf, nullptr, CM1, CD, 0, 0);
    kr_kernel<<<(CM1 * CH + 255) / 256, 256, 0, stream>>>(Kbuf, F(LB + 3), krb, CM1);
    (void)hipMemsetAsync(amax, 0, (size_t)CN * CH * 4, stream);
    (void)hipMemsetAsync(den,  0, (size_t)CN * CH * 4, stream);
    (void)hipMemsetAsync(num,  0, (size_t)CN * CD * 4, stream);
    seg_max_kernel<<<(CTOTE * CH + 255) / 256, 256, 0, stream>>>(ei0, ei1, krb, amax, 1);
    seg_sum_kernel<<<CTOTE, 256, 0, stream>>>(ei0, ei1, krb, amax, Vbuf, den, num, 1);
    ln0_kernel<<<CN, 256, 0, stream>>>(num, den, F(LB + 3), F(LB + 8), F(LB + 7),
                                       aout, aoutbf);
    gemm(aoutbf, wl + 2 * WS, F(LB + 4), ffb, nullptr, CN, CD, 0, 1);
    ln1_kernel<<<CN, 256, 0, stream>>>(aout, ffb, F(LB + 10), F(LB + 9),
                                       S_f32, S_bf, 1);                 // emb_s (relu)
  }

  // outputs: emb_s [N,D] then emb_t[:E] [E,D], f32
  (void)hipMemcpyAsync(d_out, S_f32, (size_t)CN * CD * 4, hipMemcpyDeviceToDevice, stream);
  (void)hipMemcpyAsync((float*)d_out + (size_t)CN * CD, T, (size_t)CE * CD * 4,
                       hipMemcpyDeviceToDevice, stream);
}
